// HeterogeneousGNN_41721312313370
// MI455X (gfx1250) — matrix operand order repaired and checked by reading, not compiled, not run on hardware
//
#include <hip/hip_runtime.h>

#define THREADS 256

typedef __attribute__((ext_vector_type(16))) __bf16 v16bf;
typedef __attribute__((ext_vector_type(8)))  float v8f;
typedef __attribute__((ext_vector_type(8)))  unsigned int u32x8;

__device__ __forceinline__ unsigned short f2bf(float f) {
  unsigned u = __builtin_bit_cast(unsigned, f);
  unsigned r = u + 0x7fffu + ((u >> 16) & 1u);
  return (unsigned short)(r >> 16);
}
__device__ __forceinline__ float bf2f(unsigned short h) {
  return __builtin_bit_cast(float, ((unsigned)h) << 16);
}
// low 32 bits of a flat LDS pointer == wave-relative LDS byte offset
__device__ __forceinline__ unsigned lds_off(const void* p) {
  return (unsigned)(unsigned long long)p;
}
// CDNA5 async DMA: copy 16B/lane global -> LDS, tracked by ASYNCcnt
__device__ __forceinline__ void async_copy_b128(unsigned ldsAddr, const void* base,
                                                unsigned byteOff) {
  asm volatile("global_load_async_to_lds_b128 %0, %1, %2"
               :
               : "v"(ldsAddr), "v"(byteOff), "s"(base)
               : "memory");
}
__device__ __forceinline__ void wait_async0() {
  asm volatile("s_wait_asynccnt 0" ::: "memory");
}

// ---------- copy contiguous [B][N][D] into batch-strided state ----------
__global__ __launch_bounds__(THREADS) void copy_state_kernel(
    const float* __restrict__ src, float* __restrict__ dstBase,
    int Nrows, int D, unsigned long long SB, unsigned long long total) {
  unsigned long long idx = (unsigned long long)blockIdx.x * THREADS + threadIdx.x;
  if (idx >= total) return;
  int d = (int)(idx % D);
  unsigned long long t = idx / D;
  int n = (int)(t % Nrows);
  int b = (int)(t / Nrows);
  dstBase[(unsigned long long)b * SB + (unsigned long long)n * D + d] = src[idx];
}

// ---------- adj [B][Ns][Nd] f32 -> adjT [B][Nd][Ns] bf16 ----------
__global__ __launch_bounds__(THREADS) void transpose_bf16_kernel(
    const float* __restrict__ adj, unsigned short* __restrict__ adjT, int Ns, int Nd) {
  __shared__ float tile[32][33];
  int b = blockIdx.z;
  int m0 = blockIdx.x * 32;   // Nd block
  int n0 = blockIdx.y * 32;   // Ns block
  const float* s = adj + (unsigned long long)b * Ns * Nd;
  unsigned short* dst = adjT + (unsigned long long)b * Nd * Ns;
  int tx = threadIdx.x & 31;
  int ty = threadIdx.x >> 5;  // 0..7
#pragma unroll
  for (int r = 0; r < 4; ++r)
    tile[ty + r * 8][tx] = s[(unsigned long long)(n0 + ty + r * 8) * Nd + m0 + tx];
  __syncthreads();
#pragma unroll
  for (int r = 0; r < 4; ++r) {
    int m = m0 + ty + r * 8;
    dst[(unsigned long long)m * Ns + n0 + tx] = f2bf(tile[tx][ty + r * 8]);
  }
}

// ---------- f32 -> bf16 elementwise ----------
__global__ __launch_bounds__(THREADS) void f32_to_bf16_kernel(
    const float* __restrict__ src, unsigned short* __restrict__ dst,
    unsigned long long total) {
  unsigned long long idx = (unsigned long long)blockIdx.x * THREADS + threadIdx.x;
  if (idx < total) dst[idx] = f2bf(src[idx]);
}

// ---------- v[b,:] = W_att^T @ sr[b,:]  (W is [out,in] row-major) ----------
__global__ __launch_bounds__(THREADS) void attvec_kernel(
    const float* __restrict__ W, const float* __restrict__ sr,
    float* __restrict__ v, int D) {
  __shared__ float ssr[64];
  int b = blockIdx.y;
  int i = blockIdx.x * THREADS + threadIdx.x;
  float acc = 0.f;
  for (int o0 = 0; o0 < D; o0 += 64) {
    __syncthreads();
    if (threadIdx.x < 64) ssr[threadIdx.x] = sr[b * D + o0 + threadIdx.x];
    __syncthreads();
#pragma unroll 8
    for (int oo = 0; oo < 64; ++oo)
      acc += W[(unsigned long long)(o0 + oo) * D + i] * ssr[oo];
  }
  v[b * 768 + i] = acc;
}

// ---------- c[b] = b_att . sr[b,:] ----------
__global__ __launch_bounds__(THREADS) void cdot_kernel(
    const float* __restrict__ bv, const float* __restrict__ sr,
    float* __restrict__ c, int D) {
  __shared__ float red[THREADS];
  int b = blockIdx.x;
  float acc = 0.f;
  for (int o = threadIdx.x; o < D; o += THREADS) acc += bv[o] * sr[b * D + o];
  red[threadIdx.x] = acc;
  __syncthreads();
  for (int s = THREADS / 2; s > 0; s >>= 1) {
    if (threadIdx.x < s) red[threadIdx.x] += red[threadIdx.x + s];
    __syncthreads();
  }
  if (threadIdx.x == 0) c[b] = red[0];
}

// ---------- score[b,n] = src[b,n,:].v[b,:] + c[b] (one wave per row) ----------
__global__ __launch_bounds__(THREADS) void score_kernel(
    const float* __restrict__ state, const float* __restrict__ v,
    const float* __restrict__ c, float* __restrict__ score,
    int D, unsigned long long SB) {
  int b = blockIdx.y;
  int lane = threadIdx.x & 31, wid = threadIdx.x >> 5;
  int n = blockIdx.x * 8 + wid;
  const float* row = state + (unsigned long long)b * SB + (unsigned long long)n * D;
  const float* vb = v + b * 768;
  float acc = 0.f;
  for (int d = lane; d < D; d += 32) acc += row[d] * vb[d];
#pragma unroll
  for (int off = 16; off > 0; off >>= 1) acc += __shfl_xor(acc, off, 32);
  if (lane == 0) score[b * 768 + n] = acc + c[b];
}

// ---------- softmax over n, one block per batch ----------
__global__ __launch_bounds__(THREADS) void softmax_kernel(
    const float* __restrict__ score, float* __restrict__ att, int N) {
  __shared__ float red[THREADS];
  int b = blockIdx.x;
  const float* s = score + b * 768;
  float mx = -3.0e38f;
  for (int n = threadIdx.x; n < N; n += THREADS) mx = fmaxf(mx, s[n]);
  red[threadIdx.x] = mx;
  __syncthreads();
  for (int st = THREADS / 2; st > 0; st >>= 1) {
    if (threadIdx.x < st) red[threadIdx.x] = fmaxf(red[threadIdx.x], red[threadIdx.x + st]);
    __syncthreads();
  }
  mx = red[0];
  __syncthreads();
  float sum = 0.f;
  for (int n = threadIdx.x; n < N; n += THREADS) sum += expf(s[n] - mx);
  red[threadIdx.x] = sum;
  __syncthreads();
  for (int st = THREADS / 2; st > 0; st >>= 1) {
    if (threadIdx.x < st) red[threadIdx.x] += red[threadIdx.x + st];
    __syncthreads();
  }
  float inv = 1.f / red[0];
  for (int n = threadIdx.x; n < N; n += THREADS)
    att[b * 768 + n] = expf(s[n] - mx) * inv;
}

// ---------- srcA[b,n,d] = bf16(att[b,n] * src[b,n,d]) ----------
__global__ __launch_bounds__(THREADS) void scale_src_kernel(
    const float* __restrict__ state, const float* __restrict__ att,
    unsigned short* __restrict__ srcA, int Ns, int D, unsigned long long SB,
    unsigned long long total) {
  unsigned long long idx = (unsigned long long)blockIdx.x * THREADS + threadIdx.x;
  if (idx >= total) return;
  int d = (int)(idx % D);
  unsigned long long t = idx / D;
  int n = (int)(t % Ns);
  int b = (int)(t / Ns);
  float val = att[b * 768 + n] *
              state[(unsigned long long)b * SB + (unsigned long long)n * D + d];
  srcA[idx] = f2bf(val);
}

// ---------- invd[b,m] = 1 / sum_n att[b,n]*adjT[b,m,n]  (0 -> 1) ----------
__global__ __launch_bounds__(THREADS) void denom_kernel(
    const unsigned short* __restrict__ adjT, const float* __restrict__ att,
    float* __restrict__ invd, int Nd, int Ns) {
  int b = blockIdx.y;
  int lane = threadIdx.x & 31, wid = threadIdx.x >> 5;
  int m = blockIdx.x * 8 + wid;
  const unsigned short* row =
      adjT + (unsigned long long)b * Nd * Ns + (unsigned long long)m * Ns;
  const float* ab = att + b * 768;
  float acc = 0.f;
  for (int n = lane; n < Ns; n += 32) acc += bf2f(row[n]) * ab[n];
#pragma unroll
  for (int off = 16; off > 0; off >>= 1) acc += __shfl_xor(acc, off, 32);
  if (lane == 0) invd[b * 768 + m] = (acc == 0.f) ? 1.f : 1.f / acc;
}

// =====================================================================
// WMMA GEMM 1: msgN[b,m,d] = bf16( invd[b,m] * sum_n adjT[b,m,n]*srcA[b,n,d] )
// A row-major [M,K] bf16 (async->LDS, double-buffered).
// B row-major [K,N] bf16 (regs -> transposed LDS [n][k], double-buffered).
// 128x128x32 tiles, 8 waves, LDS row stride 20 dwords (16B-aligned rows).
// =====================================================================
__global__ __launch_bounds__(THREADS) void gemm_msg_wmma(
    const unsigned short* __restrict__ A, const unsigned short* __restrict__ Bm,
    const float* __restrict__ invd, unsigned short* __restrict__ Out,
    int M, int N, int K) {
  __shared__ __align__(16) unsigned int As32[2][128 * 20];
  __shared__ __align__(16) unsigned int Bt32[2][128 * 20];
  int b = blockIdx.z;
  const unsigned int* Ag = (const unsigned int*)(A + (unsigned long long)b * M * K);
  const unsigned int* Bg = (const unsigned int*)(Bm + (unsigned long long)b * K * N);
  int tid = threadIdx.x, lane = tid & 31, wid = tid >> 5;
  int wm = wid >> 2, wn = wid & 3;          // 2 x 4 wave grid
  int m_in = lane & 15, hi = lane >> 4;
  int m0 = blockIdx.y * 128, n0 = blockIdx.x * 128;
  int Kd = K >> 1, Nd2 = N >> 1;
  unsigned asBase0 = lds_off(&As32[0][0]);          // scalar, instruction-level cast
  unsigned short* BtU = (unsigned short*)&Bt32[0][0];  // scalar generic pointer

  v8f zero = {0.f, 0.f, 0.f, 0.f, 0.f, 0.f, 0.f, 0.f};
  v8f acc[4][2];
#pragma unroll
  for (int mt = 0; mt < 4; ++mt)
#pragma unroll
    for (int nt = 0; nt < 2; ++nt) acc[mt][nt] = zero;

  auto issueA = [&](int kdw, int buf) {  // kdw = k0/2 (dword column base)
    unsigned base = asBase0 + (unsigned)buf * 10240u;  // plane = 128*20*4 bytes
#pragma unroll
    for (int r = 0; r < 2; ++r) {
      int idx = tid + r * 256;           // 0..511 quads of 16B
      int row = idx >> 2, q = idx & 3;
      async_copy_b128(base + (unsigned)(row * 80 + q * 16), Ag,
                      (unsigned)(((m0 + row) * Kd + kdw + q * 4) * 4));
    }
  };
  auto loadB = [&](int k0, unsigned int* breg) {
#pragma unroll
    for (int r = 0; r < 8; ++r) {
      int idx = tid + r * 256;
      int kk = idx >> 6, ndw = idx & 63;
      breg[r] = Bg[(unsigned long long)(k0 + kk) * Nd2 + (n0 >> 1) + ndw];
    }
  };
  auto storeB = [&](int buf, const unsigned int* breg) {
    unsigned short* bp = BtU + buf * 5120;  // plane = 128*20*2 u16
#pragma unroll
    for (int r = 0; r < 8; ++r) {
      int idx = tid + r * 256;
      int kk = idx >> 6, ndw = idx & 63;
      bp[(2 * ndw) * 40 + kk] = (unsigned short)(breg[r] & 0xffffu);
      bp[(2 * ndw + 1) * 40 + kk] = (unsigned short)(breg[r] >> 16);
    }
  };

  int nk = K >> 5;
  unsigned int breg[8];
  issueA(0, 0);
  loadB(0, breg);
  storeB(0, breg);
  wait_async0();
  __syncthreads();

  for (int kt = 0; kt < nk; ++kt) {
    int cur = kt & 1, nxt = cur ^ 1;
    bool more = (kt + 1) < nk;
    if (more) {
      issueA((kt + 1) * 16, nxt);
      loadB((kt + 1) * 32, breg);
    }
    v16bf bfrag[2];
#pragma unroll
    for (int nt = 0; nt < 2; ++nt) {
      int col = wn * 32 + nt * 16 + m_in;
      u32x8 bu;
#pragma unroll
      for (int j = 0; j < 8; ++j) bu[j] = Bt32[cur][col * 20 + j + 8 * hi];
      bfrag[nt] = __builtin_bit_cast(v16bf, bu);
    }
#pragma unroll
    for (int mt = 0; mt < 4; ++mt) {
      int row = wm * 64 + mt * 16 + m_in;
      u32x8 au;
#pragma unroll
      for (int j = 0; j < 8; ++j) {
        int kb = j + 8 * hi;
        au[j] = As32[cur][row * 20 + kb];
      }
      v16bf afrag = __builtin_bit_cast(v16bf, au);
#pragma unroll
      for (int nt = 0; nt < 2; ++nt)
        acc[mt][nt] = __builtin_amdgcn_wmma_f32_16x16x32_bf16(
            false, afrag, false, bfrag[nt], (short)0, acc[mt][nt], false, false);
    }
    if (more) storeB(nxt, breg);
    wait_async0();
    __syncthreads();
  }

  const float* ivb = invd + b * 768;
  unsigned short* ob = Out + (unsigned long long)b * M * N;
#pragma unroll
  for (int mt = 0; mt < 4; ++mt)
#pragma unroll
    for (int nt = 0; nt < 2; ++nt)
#pragma unroll
      for (int j = 0; j < 8; ++j) {
        int m = m0 + wm * 64 + mt * 16 + j + 8 * hi;
        int n = n0 + wn * 32 + nt * 16 + m_in;
        ob[(unsigned long long)m * N + n] = f2bf(acc[mt][nt][j] * ivb[m]);
      }
}

// =====================================================================
// WMMA GEMM 2: state[b,m,o] = relu( sum_i msgN[b,m,i]*W[o,i] + bias[o] + state[b,m,o] )
// A row-major [M,K] bf16 (batched), B = W bf16 [N,K] (B(k,n)=W[n*K+k]).
// Both operands streamed via async-to-LDS, double-buffered.
// =====================================================================
__global__ __launch_bounds__(THREADS) void gemm_lin_wmma(
    const unsigned short* __restrict__ A, const unsigned short* __restrict__ W,
    const float* __restrict__ bias, float* __restrict__ state,
    int M, int N, int K, unsigned long long SB) {
  __shared__ __align__(16) unsigned int As32[2][128 * 20];
  __shared__ __align__(16) unsigned int Bt32[2][128 * 20];
  int b = blockIdx.z;
  const unsigned int* Ag = (const unsigned int*)(A + (unsigned long long)b * M * K);
  const unsigned int* Wg = (const unsigned int*)W;
  int tid = threadIdx.x, lane = tid & 31, wid = tid >> 5;
  int wm = wid >> 2, wn = wid & 3;
  int m_in = lane & 15, hi = lane >> 4;
  int m0 = blockIdx.y * 128, n0 = blockIdx.x * 128;
  int Kd = K >> 1;
  unsigned asBase0 = lds_off(&As32[0][0]);
  unsigned btBase0 = lds_off(&Bt32[0][0]);

  v8f zero = {0.f, 0.f, 0.f, 0.f, 0.f, 0.f, 0.f, 0.f};
  v8f acc[4][2];
#pragma unroll
  for (int mt = 0; mt < 4; ++mt)
#pragma unroll
    for (int nt = 0; nt < 2; ++nt) acc[mt][nt] = zero;

  auto issueA = [&](int kdw, int buf) {
    unsigned base = asBase0 + (unsigned)buf * 10240u;
#pragma unroll
    for (int r = 0; r < 2; ++r) {
      int idx = tid + r * 256;
      int row = idx >> 2, q = idx & 3;
      async_copy_b128(base + (unsigned)(row * 80 + q * 16), Ag,
                      (unsigned)(((m0 + row) * Kd + kdw + q * 4) * 4));
    }
  };
  auto issueW = [&](int kdw, int buf) {
    unsigned base = btBase0 + (unsigned)buf * 10240u;
#pragma unroll
    for (int r = 0; r < 2; ++r) {
      int idx = tid + r * 256;
      int nn = idx >> 2, q = idx & 3;
      async_copy_b128(base + (unsigned)(nn * 80 + q * 16), Wg,
                      (unsigned)(((n0 + nn) * Kd + kdw + q * 4) * 4));
    }
  };

  int nk = K >> 5;
  issueA(0, 0);
  issueW(0, 0);
  wait_async0();
  __syncthreads();

  for (int kt = 0; kt < nk; ++kt) {
    int cur = kt & 1, nxt = cur ^ 1;
    bool more = (kt + 1) < nk;
    if (more) {
      issueA((kt + 1) * 16, nxt);
      issueW((kt + 1) * 16, nxt);
    }
    v16bf bfrag[2];
#pragma unroll
    for (int nt = 0; nt < 2; ++nt) {
      int col = wn * 32 + nt * 16 + m_in;
      u32x8 bu;
#pragma unroll
      for (int j = 0; j < 8; ++j) bu[j] = Bt32[cur][col * 20 + j + 8 * hi];
      bfrag[nt] = __builtin_bit_cast(v16bf, bu);
    }
#pragma unroll
    for (int mt = 0; mt < 4; ++mt) {
      int row = wm * 64 + mt * 16 + m_in;
      u32x8 au;
#pragma unroll
      for (int j = 0; j < 8; ++j) {
        int kb = j + 8 * hi;
        au[j] = As32[cur][row * 20 + kb];
      }
      v16bf afrag = __builtin_bit_cast(v16bf, au);
#pragma unroll
      for (int nt = 0; nt < 2; ++nt)
        acc[mt][nt] = __builtin_amdgcn_wmma_f32_16x16x32_bf16(
            false, afrag, false, bfrag[nt], (short)0, acc[mt][nt], false, false);
    }
    wait_async0();
    __syncthreads();
  }

  float* st = state + (unsigned long long)b * SB;
#pragma unroll
  for (int mt = 0; mt < 4; ++mt)
#pragma unroll
    for (int nt = 0; nt < 2; ++nt)
#pragma unroll
      for (int j = 0; j < 8; ++j) {
        int m = m0 + wm * 64 + mt * 16 + j + 8 * hi;
        int n = n0 + wn * 32 + nt * 16 + m_in;
        unsigned long long off = (unsigned long long)m * N + n;
        float vv = acc[mt][nt][j] + bias[n] + st[off];
        st[off] = vv > 0.f ? vv : 0.f;
      }
}

// =====================================================================
extern "C" void kernel_launch(void* const* d_in, const int* in_sizes, int n_in,
                              void* d_out, int out_size, void* d_ws, size_t ws_size,
                              hipStream_t stream) {
  (void)in_sizes; (void)n_in; (void)out_size; (void)ws_size;
  const int Bn = 32, NEV = 512, NENT = 768, D = 768, Ln = 3;
  const float* sr        = (const float*)d_in[0];
  const float* ev_in     = (const float*)d_in[1];
  const float* ent_in    = (const float*)d_in[2];
  const float* ev2ent    = (const float*)d_in[3];
  const float* ent2ev    = (const float*)d_in[4];
  const float* W_ev_att  = (const float*)d_in[5];
  const float* b_ev_att  = (const float*)d_in[6];
  const float* W_ev_ent  = (const float*)d_in[7];
  const float* b_ev_ent  = (const float*)d_in[8];
  const float* W_ent_att = (const float*)d_in[9];
  const float* b_ent_att = (const float*)d_in[10];
  const float* W_ent_ev  = (const float*)d_in[11];
  const float* b_ent_ev  = (const float*)d_in[12];

  const unsigned long long SB = (unsigned long long)(NENT + NEV) * D; // d_out batch stride
  float* out = (float*)d_out;
  float* ent_state = out;                                  // [b][0..NENT)[D]
  float* ev_state  = out + (unsigned long long)NENT * D;   // [b][NENT..NENT+NEV)[D]

  char* ws = (char*)d_ws;
  auto carve = [&](size_t bytes) -> void* {
    void* p = (void*)ws;
    ws += (bytes + 255) & ~(size_t)255;
    return p;
  };
  unsigned short* adjT_e2n  = (unsigned short*)carve((size_t)Bn * NENT * NEV * 2);
  unsigned short* adjT_n2e  = (unsigned short*)carve((size_t)Bn * NEV * NENT * 2);
  unsigned short* Wbf_event = (unsigned short*)carve((size_t)Ln * D * D * 2);
  unsigned short* Wbf_entev = (unsigned short*)carve((size_t)Ln * D * D * 2);
  unsigned short* srcA      = (unsigned short*)carve((size_t)Bn * NENT * D * 2);
  unsigned short* msgN      = (unsigned short*)carve((size_t)Bn * NENT * D * 2);
  float* vbuf  = (float*)carve((size_t)Bn * 768 * 4);
  float* cbuf  = (float*)carve((size_t)Bn * 4);
  float* score = (float*)carve((size_t)Bn * 768 * 4);
  float* att   = (float*)carve((size_t)Bn * 768 * 4);
  float* invd  = (float*)carve((size_t)Bn * 768 * 4);

  // --- one-time prep ---
  {
    unsigned long long tot = (unsigned long long)Bn * NENT * D;
    copy_state_kernel<<<(unsigned)((tot + THREADS - 1) / THREADS), THREADS, 0, stream>>>(
        ent_in, ent_state, NENT, D, SB, tot);
  }
  {
    unsigned long long tot = (unsigned long long)Bn * NEV * D;
    copy_state_kernel<<<(unsigned)((tot + THREADS - 1) / THREADS), THREADS, 0, stream>>>(
        ev_in, ev_state, NEV, D, SB, tot);
  }
  transpose_bf16_kernel<<<dim3(NENT / 32, NEV / 32, Bn), THREADS, 0, stream>>>(
      ev2ent, adjT_e2n, NEV, NENT);
  transpose_bf16_kernel<<<dim3(NEV / 32, NENT / 32, Bn), THREADS, 0, stream>>>(
      ent2ev, adjT_n2e, NENT, NEV);
  {
    unsigned long long tot = (unsigned long long)Ln * D * D;
    unsigned nb = (unsigned)((tot + THREADS - 1) / THREADS);
    f32_to_bf16_kernel<<<nb, THREADS, 0, stream>>>(W_ev_ent, Wbf_event, tot);
    f32_to_bf16_kernel<<<nb, THREADS, 0, stream>>>(W_ent_ev, Wbf_entev, tot);
  }

  auto run_pass = [&](const float* src_state, int Ns, float* dst_state, int Nd,
                      const unsigned short* adjT, const float* Watt,
                      const float* batt, const unsigned short* Wmsg,
                      const float* bmsg) {
    attvec_kernel<<<dim3(D / THREADS, Bn), THREADS, 0, stream>>>(Watt, sr, vbuf, D);
    cdot_kernel<<<Bn, THREADS, 0, stream>>>(batt, sr, cbuf, D);
    score_kernel<<<dim3(Ns / 8, Bn), THREADS, 0, stream>>>(src_state, vbuf, cbuf,
                                                           score, D, SB);
    softmax_kernel<<<Bn, THREADS, 0, stream>>>(score, att, Ns);
    {
      unsigned long long tot = (unsigned long long)Bn * Ns * D;
      scale_src_kernel<<<(unsigned)((tot + THREADS - 1) / THREADS), THREADS, 0,
                         stream>>>(src_state, att, srcA, Ns, D, SB, tot);
    }
    denom_kernel<<<dim3(Nd / 8, Bn), THREADS, 0, stream>>>(adjT, att, invd, Nd, Ns);
    gemm_msg_wmma<<<dim3(D / 128, Nd / 128, Bn), THREADS, 0, stream>>>(
        adjT, srcA, invd, msgN, Nd, D, Ns);
    gemm_lin_wmma<<<dim3(D / 128, Nd / 128, Bn), THREADS, 0, stream>>>(
        msgN, Wmsg, bmsg, dst_state, Nd, D, D, SB);
  };

  for (int i = 0; i < Ln; ++i) {
    // evidence -> entity
    run_pass(ev_state, NEV, ent_state, NENT, adjT_e2n,
             W_ev_att + (size_t)i * D * D, b_ev_att + i * D,
             Wbf_event + (size_t)i * D * D, b_ev_ent + i * D);
    // entity -> evidence
    run_pass(ent_state, NENT, ev_state, NEV, adjT_n2e,
             W_ent_att + (size_t)i * D * D, b_ent_att + i * D,
             Wbf_entev + (size_t)i * D * D, b_ent_ev + i * D);
  }
}
